// ConvaiCausalLMAttention_28441273434743
// MI455X (gfx1250) — compile-verified, matrix operand order checked by reading, not run on hardware
//
#include <hip/hip_runtime.h>
#include <hip/hip_bf16.h>

// MI455X (gfx1250) causal attention layer.
//   - One pass converts X and all weights f32->bf16 (bandwidth-trivial).
//   - NT GEMMs (x @ W^T) on v_wmma_f32_16x16x32_bf16, double-buffered LDS
//     tiles fed by the Tensor Data Mover (tensor_load_to_lds + tensorcnt).
//   - Flash attention with online softmax; K tiles via TDM, V transposed
//     manually into LDS, P staged through LDS to reach WMMA A-layout.
// Problem is compute-bound (~1.5 kFLOP/byte); all operands fit in 192MB L2.
// Workspace use: ~76 MB bf16 staging (see kernel_launch).

#define HDIM   2048
#define SEQ    4096
#define NHEADS 32
#define NKVH   8
#define HEADD  64
#define KVW    (NKVH * HEADD)   // 512

typedef __attribute__((ext_vector_type(16))) __bf16 v16bf;
typedef __attribute__((ext_vector_type(8)))  float  v8f;
typedef __attribute__((ext_vector_type(4)))  unsigned int u32x4;
typedef __attribute__((ext_vector_type(8)))  int i32x8;
typedef __attribute__((ext_vector_type(4)))  int i32x4;

static __device__ __forceinline__ v16bf ld_frag16(const __bf16* p) {
  union U { v16bf v; uint4 q[2]; } u;
  u.q[0] = *(const uint4*)(p);
  u.q[1] = *(const uint4*)(p + 8);
  return u.v;
}

static __device__ __forceinline__ v8f wmma_bf16(v16bf a, v16bf b, v8f c) {
  return __builtin_amdgcn_wmma_f32_16x16x32_bf16(false, a, false, b, (short)0, c,
                                                 false, false);
}

static __device__ __forceinline__ v8f zero8() {
  v8f z = {0.f, 0.f, 0.f, 0.f, 0.f, 0.f, 0.f, 0.f};
  return z;
}

// ---- Tensor Data Mover: load a (tile_rows x tile_k) bf16 tile into LDS ----
// gtile points at the tile's first element; rows are rowlen_elems apart.
// LDS rows are padded: pad_interval_code/pad_amount_code per D# group1 spec.
static __device__ __forceinline__ void tdm_load_tile(
    const void* gtile, unsigned lds_byte_off,
    unsigned rowlen_elems, unsigned totrows,
    unsigned tile_k, unsigned tile_rows,
    unsigned pad_interval_code, unsigned pad_amount_code)
{
  const unsigned long long ga = (unsigned long long)(uintptr_t)gtile;

  u32x4 g0;
  g0[0] = 1u;                                     // count=1 (valid), user mode
  g0[1] = lds_byte_off;                           // lds_addr
  g0[2] = (unsigned)ga;                           // global_addr[31:0]
  g0[3] = (unsigned)((ga >> 32) & 0x01FFFFFFu)    // global_addr[56:32]
        | (2u << 30);                             // type = 2 ("image")

  i32x8 g1;
  g1[0] = (int)((1u << 16)                        // data_size = 2 bytes
        | (1u << 20)                              // pad_enable
        | (pad_interval_code << 22)               // dwords-before-pad code
        | (pad_amount_code << 25));               // pad dwords code
  g1[1] = (int)((rowlen_elems & 0xFFFFu) << 16);  // tensor_dim0[15:0]
  g1[2] = (int)(((rowlen_elems >> 16) & 0xFFFFu)  // tensor_dim0[31:16]
        | ((totrows & 0xFFFFu) << 16));           // tensor_dim1[15:0]
  g1[3] = (int)(((totrows >> 16) & 0xFFFFu)       // tensor_dim1[31:16]
        | ((tile_k & 0xFFFFu) << 16));            // tile_dim0
  g1[4] = (int)(tile_rows & 0xFFFFu);             // tile_dim1 (tile_dim2 = 0)
  g1[5] = (int)rowlen_elems;                      // tensor_dim0_stride[31:0]
  g1[6] = 0;                                      // stride[47:32] | dim1_stride lo
  g1[7] = 0;

  i32x4 g2 = {0, 0, 0, 0};
  i32x4 g3 = {0, 0, 0, 0};

#if __clang_major__ >= 23
  i32x8 z8 = {0, 0, 0, 0, 0, 0, 0, 0};
  __builtin_amdgcn_tensor_load_to_lds(g0, g1, g2, g3, z8, 0);
#else
  __builtin_amdgcn_tensor_load_to_lds(g0, g1, g2, g3, 0);
#endif
}

// ---- f32 -> bf16 bulk convert (8 elements / thread) ----
__global__ __launch_bounds__(256) void cvt_f32_bf16(
    const float* __restrict__ s, __bf16* __restrict__ d, long n)
{
  const long i = ((long)blockIdx.x * 256 + threadIdx.x) * 8;
  if (i >= n) return;
  float4 a = *(const float4*)(s + i);
  float4 b = *(const float4*)(s + i + 4);
  union { uint4 q; __bf16 h[8]; } u;
  u.h[0] = (__bf16)a.x; u.h[1] = (__bf16)a.y;
  u.h[2] = (__bf16)a.z; u.h[3] = (__bf16)a.w;
  u.h[4] = (__bf16)b.x; u.h[5] = (__bf16)b.y;
  u.h[6] = (__bf16)b.z; u.h[7] = (__bf16)b.w;
  *(uint4*)(d + i) = u.q;
}

// ---- C[M,N] = A[M,K] * B[N,K]^T, bf16 WMMA, TDM-fed double-buffered LDS ----
// Block tile 128x128, K-step 32, 256 threads = 8 waves (4x2), wave tile 32x64.
template <typename OT>
__global__ __launch_bounds__(256) void gemm_nt_tdm(
    const __bf16* __restrict__ A, const __bf16* __restrict__ B,
    OT* __restrict__ C, int M, int N, int K)
{
  __shared__ __bf16 As[2][128][40];   // 32-half rows + 8-half pad (TDM pads)
  __shared__ __bf16 Bs[2][128][40];

  const int tid  = threadIdx.x;
  const int lane = tid & 31;
  const int wave = tid >> 5;
  const int l15  = lane & 15;
  const int kh   = (lane >> 4) * 16;

  const int m0 = blockIdx.y * 128;
  const int n0 = blockIdx.x * 128;
  const int m0w = (wave & 3) * 32;
  const int n0w = (wave >> 2) * 64;

  v8f acc[2][4];
  #pragma unroll
  for (int i = 0; i < 2; ++i)
    #pragma unroll
    for (int j = 0; j < 4; ++j) acc[i][j] = zero8();

  const int nk = K >> 5;   // K/32 steps
  // Prologue: DMA tile 0 into buffer 0 (wave 0 issues; EXEC-independent DMA).
  if (tid < 32) {
    // LDS row = 32 halves = 16 dwords -> interval code 3; pad 4 dwords -> code 3
    tdm_load_tile(A + (size_t)m0 * K, (unsigned)(uintptr_t)&As[0][0][0],
                  (unsigned)K, (unsigned)M, 32u, 128u, 3u, 3u);
    tdm_load_tile(B + (size_t)n0 * K, (unsigned)(uintptr_t)&Bs[0][0][0],
                  (unsigned)K, (unsigned)N, 32u, 128u, 3u, 3u);
  }

  for (int i = 0; i < nk; ++i) {
    if (tid < 32) __builtin_amdgcn_s_wait_tensorcnt(0);
    __syncthreads();   // tile i resident; all waves done with buffer i&1^1

    if (tid < 32 && (i + 1) < nk) {  // prefetch tile i+1 while computing i
      const int b1 = (i + 1) & 1;
      tdm_load_tile(A + (size_t)m0 * K + (i + 1) * 32,
                    (unsigned)(uintptr_t)&As[b1][0][0],
                    (unsigned)K, (unsigned)M, 32u, 128u, 3u, 3u);
      tdm_load_tile(B + (size_t)n0 * K + (i + 1) * 32,
                    (unsigned)(uintptr_t)&Bs[b1][0][0],
                    (unsigned)K, (unsigned)N, 32u, 128u, 3u, 3u);
    }

    const int b = i & 1;
    v16bf af[2];
    #pragma unroll
    for (int mt = 0; mt < 2; ++mt)
      af[mt] = ld_frag16(&As[b][m0w + mt * 16 + l15][kh]);
    #pragma unroll
    for (int nt = 0; nt < 4; ++nt) {
      v16bf bfrag = ld_frag16(&Bs[b][n0w + nt * 16 + l15][kh]);
      #pragma unroll
      for (int mt = 0; mt < 2; ++mt)
        acc[mt][nt] = wmma_bf16(af[mt], bfrag, acc[mt][nt]);
    }
  }

  const int radd = (lane >> 4) * 8;
  #pragma unroll
  for (int mt = 0; mt < 2; ++mt)
    #pragma unroll
    for (int nt = 0; nt < 4; ++nt)
      #pragma unroll
      for (int r = 0; r < 8; ++r) {
        const int m = m0 + m0w + mt * 16 + r + radd;
        const int n = n0 + n0w + nt * 16 + l15;
        C[(size_t)m * N + n] = (OT)acc[mt][nt][r];
      }
}

// ---- Flash attention: 64 q-rows x head per block; 128 threads = 4 waves ----
__global__ __launch_bounds__(128) void flash_attn_kernel(
    const __bf16* __restrict__ Q, const __bf16* __restrict__ K,
    const __bf16* __restrict__ V, __bf16* __restrict__ O)
{
  __shared__ __bf16 Ks[64][72];        // K tile [j][d], TDM-filled (64+8 pad)
  __shared__ __bf16 Vt[64][72];        // V tile transposed [d][j]
  __shared__ __bf16 Ps[4][16][72];     // per-wave P staging (C -> A layout)

  const int tid  = threadIdx.x;
  const int lane = tid & 31;
  const int wave = tid >> 5;
  const int l15  = lane & 15;
  const int hh   = lane >> 4;
  const int qb   = blockIdx.x;
  const int h    = blockIdx.y;
  const int kvh  = h >> 2;             // GROUPS = 4
  const int qr0  = qb * 64 + wave * 16;

  v16bf qf[2];
  #pragma unroll
  for (int kc = 0; kc < 2; ++kc)
    qf[kc] = ld_frag16(Q + (size_t)(qr0 + l15) * HDIM + h * HEADD + kc * 32 + hh * 16);

  v8f oacc[4];
  float mstate[8], lstate[8];
  #pragma unroll
  for (int dt = 0; dt < 4; ++dt) oacc[dt] = zero8();
  #pragma unroll
  for (int r = 0; r < 8; ++r) { mstate[r] = -1e30f; lstate[r] = 0.f; }

  for (int kb = 0; kb <= qb; ++kb) {
    // ---- K tile via TDM (row = 64 halves = 32 dw -> code 4; pad 4 dw -> 3)
    if (tid < 32)
      tdm_load_tile(K + (size_t)(kb * 64) * KVW + kvh * HEADD,
                    (unsigned)(uintptr_t)&Ks[0][0],
                    KVW, SEQ, 64u, 64u, 4u, 3u);
    // ---- V tile, transposed into LDS by all threads
    #pragma unroll
    for (int i = 0; i < 4; ++i) {
      const int id  = tid + i * 128;
      const int row = id >> 3;
      const int g   = (id & 7) * 8;
      union { uint4 q; __bf16 hv[8]; } vu;
      vu.q = *(const uint4*)(V + (size_t)(kb * 64 + row) * KVW + kvh * HEADD + g);
      #pragma unroll
      for (int j = 0; j < 8; ++j) Vt[g + j][row] = vu.hv[j];
    }
    if (tid < 32) __builtin_amdgcn_s_wait_tensorcnt(0);
    __syncthreads();

    // ---- S = Q K^T (scaled)
    float sv[4][8];
    #pragma unroll
    for (int nt = 0; nt < 4; ++nt) {
      v8f s = zero8();
      #pragma unroll
      for (int kc = 0; kc < 2; ++kc) {
        v16bf kf = ld_frag16(&Ks[nt * 16 + l15][kc * 32 + hh * 16]);
        s = wmma_bf16(qf[kc], kf, s);
      }
      #pragma unroll
      for (int r = 0; r < 8; ++r) sv[nt][r] = s[r] * 0.125f;  // 1/sqrt(64)
    }
    if (kb == qb) {  // causal mask on the diagonal block only
      #pragma unroll
      for (int nt = 0; nt < 4; ++nt) {
        const int jcol = kb * 64 + nt * 16 + l15;
        #pragma unroll
        for (int r = 0; r < 8; ++r) {
          const int irow = qr0 + r + hh * 8;
          if (jcol > irow) sv[nt][r] = -1e30f;
        }
      }
    }

    // ---- online softmax (row stats redundant across 16-lane halves)
    #pragma unroll
    for (int r = 0; r < 8; ++r) {
      float pm = fmaxf(fmaxf(sv[0][r], sv[1][r]), fmaxf(sv[2][r], sv[3][r]));
      #pragma unroll
      for (int msk = 1; msk < 16; msk <<= 1) pm = fmaxf(pm, __shfl_xor(pm, msk, 32));
      const float mnew  = fmaxf(mstate[r], pm);
      const float alpha = __expf(mstate[r] - mnew);
      float rs = 0.f;
      #pragma unroll
      for (int nt = 0; nt < 4; ++nt) {
        const float p = __expf(sv[nt][r] - mnew);
        sv[nt][r] = p;
        rs += p;
      }
      #pragma unroll
      for (int msk = 1; msk < 16; msk <<= 1) rs += __shfl_xor(rs, msk, 32);
      lstate[r] = lstate[r] * alpha + rs;
      mstate[r] = mnew;
      #pragma unroll
      for (int dt = 0; dt < 4; ++dt) oacc[dt][r] = oacc[dt][r] * alpha;
    }

    // ---- P: C-layout -> LDS -> A-layout (intra-wave DS is in-order)
    #pragma unroll
    for (int nt = 0; nt < 4; ++nt)
      #pragma unroll
      for (int r = 0; r < 8; ++r)
        Ps[wave][r + hh * 8][nt * 16 + l15] = (__bf16)sv[nt][r];

    // ---- O += P * V
    #pragma unroll
    for (int kc = 0; kc < 2; ++kc) {
      v16bf pf = ld_frag16(&Ps[wave][l15][kc * 32 + hh * 16]);
      #pragma unroll
      for (int dt = 0; dt < 4; ++dt) {
        v16bf vf = ld_frag16(&Vt[dt * 16 + l15][kc * 32 + hh * 16]);
        oacc[dt] = wmma_bf16(pf, vf, oacc[dt]);
      }
    }
    __syncthreads();
  }

  #pragma unroll
  for (int r = 0; r < 8; ++r) {
    const float inv = 1.0f / lstate[r];
    const int row = qr0 + r + hh * 8;
    #pragma unroll
    for (int dt = 0; dt < 4; ++dt) {
      const int d = dt * 16 + l15;
      O[(size_t)row * HDIM + h * HEADD + d] = (__bf16)(oacc[dt][r] * inv);
    }
  }
}

extern "C" void kernel_launch(void* const* d_in, const int* in_sizes, int n_in,
                              void* d_out, int out_size, void* d_ws, size_t ws_size,
                              hipStream_t stream) {
  const float* X  = (const float*)d_in[0];
  // d_in[1] = additive causal mask: applied analytically in flash_attn_kernel
  const float* Wq = (const float*)d_in[2];
  const float* Wk = (const float*)d_in[3];
  const float* Wv = (const float*)d_in[4];
  const float* Wo = (const float*)d_in[5];
  float* out = (float*)d_out;

  // bf16 workspace (~76 MB): X | Wq | Wk | Wv | Wo | Q | K | V | attn-out
  __bf16* Xb  = (__bf16*)d_ws;
  __bf16* Wqb = Xb  + (size_t)SEQ * HDIM;
  __bf16* Wkb = Wqb + (size_t)HDIM * HDIM;
  __bf16* Wvb = Wkb + (size_t)KVW * HDIM;
  __bf16* Wob = Wvb + (size_t)KVW * HDIM;
  __bf16* Qb  = Wob + (size_t)HDIM * HDIM;
  __bf16* Kb  = Qb  + (size_t)SEQ * HDIM;
  __bf16* Vb  = Kb  + (size_t)SEQ * KVW;
  __bf16* Ob  = Vb  + (size_t)SEQ * KVW;

  const long nX = (long)SEQ * HDIM, nWq = (long)HDIM * HDIM, nWk = (long)KVW * HDIM;
  cvt_f32_bf16<<<nX  / 2048, 256, 0, stream>>>(X,  Xb,  nX);
  cvt_f32_bf16<<<nWq / 2048, 256, 0, stream>>>(Wq, Wqb, nWq);
  cvt_f32_bf16<<<nWk / 2048, 256, 0, stream>>>(Wk, Wkb, nWk);
  cvt_f32_bf16<<<nWk / 2048, 256, 0, stream>>>(Wv, Wvb, nWk);
  cvt_f32_bf16<<<nWq / 2048, 256, 0, stream>>>(Wo, Wob, nWq);

  gemm_nt_tdm<__bf16>
      <<<dim3(HDIM / 128, SEQ / 128), 256, 0, stream>>>(Xb, Wqb, Qb, SEQ, HDIM, HDIM);
  gemm_nt_tdm<__bf16>
      <<<dim3(KVW / 128, SEQ / 128), 256, 0, stream>>>(Xb, Wkb, Kb, SEQ, KVW, HDIM);
  gemm_nt_tdm<__bf16>
      <<<dim3(KVW / 128, SEQ / 128), 256, 0, stream>>>(Xb, Wvb, Vb, SEQ, KVW, HDIM);

  flash_attn_kernel<<<dim3(SEQ / 64, NHEADS), 128, 0, stream>>>(Qb, Kb, Vb, Ob);

  gemm_nt_tdm<float>
      <<<dim3(HDIM / 128, SEQ / 128), 256, 0, stream>>>(Ob, Wob, out, SEQ, HDIM, HDIM);
}
